// GRULWAN_34239479284347
// MI455X (gfx1250) — compile-verified, matrix operand order checked by reading, not run on hardware
//
#include <hip/hip_runtime.h>
#include <hip/hip_bf16.h>

// ---------------------------------------------------------------------------
// MI455X (gfx1250) HAN GRU + label-wise attention.
// All heavy math through V_WMMA_F32_16X16X32_BF16 (wave32 WMMA).
// Tile staging uses GLOBAL_LOAD_ASYNC_TO_LDS_B128 (ASYNCcnt) when the builtin
// is available, else plain b128 copies. Both WMMA operands are staged
// row-major (B columns == rows of N x K weight matrices), so every fragment
// load is two contiguous ds_load_b128 — no scalar LDS packing.
// ---------------------------------------------------------------------------

typedef __attribute__((ext_vector_type(16))) __bf16 v16bf;
typedef __attribute__((ext_vector_type(8)))  __bf16 v8bf;
typedef __attribute__((ext_vector_type(8)))  float  v8f;
typedef __attribute__((ext_vector_type(4)))  int    v4i;

#if defined(__HIP_DEVICE_COMPILE__) && \
    __has_builtin(__builtin_amdgcn_global_load_async_to_lds_b128)
#define USE_ASYNC_LDS 1
#else
#define USE_ASYNC_LDS 0
#endif

// 16-byte global -> LDS stage (async data mover when available).
// Builtin signature (from clang diagnostic): (v4i addrspace(1)*, v4i
// addrspace(3)*, imm offset, imm cpol).
__device__ __forceinline__ void lds_cp16(__bf16* dst, const __bf16* src) {
#if USE_ASYNC_LDS
  __builtin_amdgcn_global_load_async_to_lds_b128(
      (__attribute__((address_space(1))) v4i*)(src),
      (__attribute__((address_space(3))) v4i*)(dst), 0, 0);
#else
  *(v8bf*)dst = *(const v8bf*)src;
#endif
}
__device__ __forceinline__ void lds_stage_wait() {
#if USE_ASYNC_LDS
#if __has_builtin(__builtin_amdgcn_s_wait_asynccnt)
  __builtin_amdgcn_s_wait_asynccnt(0);
#else
  asm volatile("s_wait_asynccnt 0" ::: "memory");
#endif
#endif
}

// ---- WMMA fragment loader (shared by A and B operands) ---------------------
// A 16xK: lane L<16 -> row L, K = 0..7 & 16..23 ; lane L>=16 -> row L-16,
// K = 8..15 & 24..31. B Kx16 is the same pattern with "row" -> "column", and
// our B tiles are staged row-major as W[n][k], so one helper serves both.
__device__ __forceinline__ v16bf load_frag_row(const __bf16* base, int ld, int r0, int lane) {
  const __bf16* p = base + (size_t)(r0 + (lane & 15)) * ld + ((lane & 16) ? 8 : 0);
  v16bf f;
#pragma unroll
  for (int e = 0; e < 8; ++e) f[e] = p[e];
#pragma unroll
  for (int e = 0; e < 8; ++e) f[8 + e] = p[16 + e];
  return f;
}

// ---------------------------------------------------------------------------
// Generic WMMA GEMM: C[M,N] = A[M,K](bf16 row-major) x Bt[N,K](bf16)^T + bias
// Block: 256 threads (8 waves, 2x4), tile 128x128, K-step 32.
// ---------------------------------------------------------------------------
template <typename OutT>
__global__ void __launch_bounds__(256)
wmma_gemm_bt(const __bf16* __restrict__ A, const __bf16* __restrict__ Bt,
             const float* __restrict__ bias, OutT* __restrict__ C,
             int M, int N, int K) {
  __shared__ __align__(16) __bf16 As[128 * 32];
  __shared__ __align__(16) __bf16 Ws[128 * 32];
  const int tid = threadIdx.x, lane = tid & 31, wave = tid >> 5;
  const int wm = wave >> 2, wn = wave & 3;
  const int m0 = blockIdx.x * 128, n0 = blockIdx.y * 128;
  v8f acc[4][2] = {};
  for (int k0 = 0; k0 < K; k0 += 32) {
#pragma unroll
    for (int p = 0; p < 2; ++p) {                 // stage both tiles row-major
      int v = tid + p * 256;
      int row = v >> 2, kc = (v & 3) << 3;
      lds_cp16(As + row * 32 + kc, A + (size_t)(m0 + row) * K + k0 + kc);
      lds_cp16(Ws + row * 32 + kc, Bt + (size_t)(n0 + row) * K + k0 + kc);
    }
    if (k0 + 32 < K) {                            // global_prefetch_b8 next tile
      __builtin_prefetch(A + (size_t)(m0 + (tid >> 1)) * K + k0 + 32, 0, 0);
      __builtin_prefetch(Bt + (size_t)(n0 + (tid >> 1)) * K + k0 + 32, 0, 0);
    }
    lds_stage_wait();
    __syncthreads();
    v16bf bf0 = load_frag_row(Ws, 32, wn * 32, lane);
    v16bf bf1 = load_frag_row(Ws, 32, wn * 32 + 16, lane);
#pragma unroll
    for (int mi = 0; mi < 4; ++mi) {
      v16bf af = load_frag_row(As, 32, wm * 64 + mi * 16, lane);
      acc[mi][0] = __builtin_amdgcn_wmma_f32_16x16x32_bf16(
          false, af, false, bf0, (short)0, acc[mi][0], false, false);
      acc[mi][1] = __builtin_amdgcn_wmma_f32_16x16x32_bf16(
          false, af, false, bf1, (short)0, acc[mi][1], false, false);
    }
    __syncthreads();
  }
  // C/D layout: lanes 0-15 -> N=lane, M=v; lanes 16-31 -> N=lane-16, M=v+8
#pragma unroll
  for (int mi = 0; mi < 4; ++mi) {
    int rb = m0 + wm * 64 + mi * 16 + ((lane >> 4) << 3);
#pragma unroll
    for (int ni = 0; ni < 2; ++ni) {
      int col = n0 + wn * 32 + ni * 16 + (lane & 15);
      float bv = bias ? bias[col] : 0.0f;
#pragma unroll
      for (int v = 0; v < 8; ++v)
        C[(size_t)(rb + v) * N + col] = (OutT)(acc[mi][ni][v] + bv);
    }
  }
}

// ---------------------------------------------------------------------------
// GRU scan. grid = (8 batch chunks of 32, 2 directions), block = 256 (8 waves)
// LDS: Whh (384x128 bf16, row-major g x k) + h (32x128 bf16) + gh (32x384 f32)
// Each step: gh = h x Whh^T via 24 WMMAs, then fused gate math.
// ---------------------------------------------------------------------------
__global__ void __launch_bounds__(256)
gru_scan_kernel(const __bf16* __restrict__ giF, const __bf16* __restrict__ giB,
                const __bf16* __restrict__ WhhFb, const __bf16* __restrict__ WhhBb,
                const float* __restrict__ bhhF, const float* __restrict__ bhhB,
                __bf16* __restrict__ H) {
  extern __shared__ __align__(16) char smem[];
  __bf16* sWhh = (__bf16*)smem;                                  // [384][128]
  __bf16* sh   = (__bf16*)(smem + 384 * 128 * 2);                // [32][128]
  float*  sgh  = (float*)(smem + 384 * 128 * 2 + 32 * 128 * 2);  // [32][384]
  const int tid = threadIdx.x, lane = tid & 31, wave = tid >> 5;
  const int dir = blockIdx.y;
  const int n0 = blockIdx.x * 32;
  const __bf16* gi  = dir ? giB : giF;
  const __bf16* Whh = dir ? WhhBb : WhhFb;
  const float*  bhh = dir ? bhhB : bhhF;

  for (int i = tid * 8; i < 384 * 128; i += 256 * 8)
    *(v8bf*)(sWhh + i) = *(const v8bf*)(Whh + i);
  for (int i = tid; i < 32 * 128; i += 256) sh[i] = (__bf16)0.0f;
  __syncthreads();

  for (int s = 0; s < 128; ++s) {
    const int t = dir ? (127 - s) : s;
    v8f acc[2][3] = {};
#pragma unroll
    for (int kt = 0; kt < 4; ++kt) {              // K = 128, 32 per WMMA
      v16bf a0 = load_frag_row(sh + kt * 32, 128, 0, lane);
      v16bf a1 = load_frag_row(sh + kt * 32, 128, 16, lane);
#pragma unroll
      for (int j = 0; j < 3; ++j) {               // 24 N-tiles / 8 waves
        v16bf b = load_frag_row(sWhh + kt * 32, 128, (wave * 3 + j) * 16, lane);
        acc[0][j] = __builtin_amdgcn_wmma_f32_16x16x32_bf16(
            false, a0, false, b, (short)0, acc[0][j], false, false);
        acc[1][j] = __builtin_amdgcn_wmma_f32_16x16x32_bf16(
            false, a1, false, b, (short)0, acc[1][j], false, false);
      }
    }
#pragma unroll
    for (int mi = 0; mi < 2; ++mi) {
      int rb = mi * 16 + ((lane >> 4) << 3);
#pragma unroll
      for (int j = 0; j < 3; ++j) {
        int col = (wave * 3 + j) * 16 + (lane & 15);
#pragma unroll
        for (int v = 0; v < 8; ++v) sgh[(rb + v) * 384 + col] = acc[mi][j][v];
      }
    }
    __syncthreads();
    // fused gates (PyTorch order r,z,n): 32x128 elems, 16 per thread
    for (int i = tid; i < 32 * 128; i += 256) {
      int r = i >> 7, j = i & 127;
      float hr = sgh[r * 384 + j]       + bhh[j];
      float hz = sgh[r * 384 + 128 + j] + bhh[128 + j];
      float hn = sgh[r * 384 + 256 + j] + bhh[256 + j];
      const __bf16* gp = gi + ((size_t)t * 256 + n0 + r) * 384;
      float ir = (float)gp[j], iz = (float)gp[128 + j], in = (float)gp[256 + j];
      float rg = 1.0f / (1.0f + __expf(-(ir + hr)));
      float zg = 1.0f / (1.0f + __expf(-(iz + hz)));
      float ng = tanhf(in + rg * hn);
      float hprev = (float)sh[r * 128 + j];
      float hnew = (1.0f - zg) * ng + zg * hprev;
      __bf16 hb = (__bf16)hnew;
      sh[r * 128 + j] = hb;
      // H layout: row = n*128 + t, 256 cols (fwd cols 0..127, bwd 128..255)
      H[((size_t)(n0 + r) * 128 + t) * 256 + dir * 128 + j] = hb;
    }
    __syncthreads();
  }
}

// ---------------------------------------------------------------------------
// Softmax over dim b (size 128) of logits (n*128+b, c) -- in place, online.
// ---------------------------------------------------------------------------
__global__ void __launch_bounds__(256)
softmax_b_kernel(float* __restrict__ L) {
  int id = blockIdx.x * blockDim.x + threadIdx.x;
  if (id >= 256 * 1024) return;
  int n = id >> 10, c = id & 1023;
  float* base = L + (size_t)n * 128 * 1024 + c;
  float m = -3.0e38f, ssum = 0.0f;
  for (int b = 0; b < 128; ++b) {
    float v = base[(size_t)b * 1024];
    if (v > m) { ssum = ssum * __expf(m - v) + 1.0f; m = v; }
    else        ssum += __expf(v - m);
  }
  float inv = 1.0f / ssum;
  for (int b = 0; b < 128; ++b) {
    float v = base[(size_t)b * 1024];
    base[(size_t)b * 1024] = __expf(v - m) * inv;
  }
}

// ---------------------------------------------------------------------------
// Fused attention output: y[b,c] = sum_n A[n,b,c] * (H[n,b,:].ow2[c,:]) + bias_c
// grid = (8 c-tiles, 128 b). G tiles computed by WMMA, reduced immediately.
// ---------------------------------------------------------------------------
__global__ void __launch_bounds__(256)
attn_out_kernel(const __bf16* __restrict__ Hbf, const __bf16* __restrict__ ow2,
                const float* __restrict__ Aout, const float* __restrict__ biasc,
                float* __restrict__ y) {
  __shared__ __align__(16) __bf16 As[128 * 32];
  __shared__ __align__(16) __bf16 Ws[128 * 32];
  __shared__ float yred[2][128];
  const int tid = threadIdx.x, lane = tid & 31, wave = tid >> 5;
  const int wm = wave >> 2, wn = wave & 3;
  const int b = blockIdx.y, c0 = blockIdx.x * 128;
  yred[tid >> 7][tid & 127] = 0.0f;
  __syncthreads();

  for (int n0 = 0; n0 < 256; n0 += 128) {
    v8f acc[4][2] = {};
    for (int k0 = 0; k0 < 256; k0 += 32) {
#pragma unroll
      for (int p = 0; p < 2; ++p) {
        int v = tid + p * 256;
        int row = v >> 2, kc = (v & 3) << 3;
        lds_cp16(As + row * 32 + kc,
                 Hbf + ((size_t)(n0 + row) * 128 + b) * 256 + k0 + kc);
        lds_cp16(Ws + row * 32 + kc,
                 ow2 + (size_t)(c0 + row) * 256 + k0 + kc);
      }
      lds_stage_wait();
      __syncthreads();
      v16bf bf0 = load_frag_row(Ws, 32, wn * 32, lane);
      v16bf bf1 = load_frag_row(Ws, 32, wn * 32 + 16, lane);
#pragma unroll
      for (int mi = 0; mi < 4; ++mi) {
        v16bf af = load_frag_row(As, 32, wm * 64 + mi * 16, lane);
        acc[mi][0] = __builtin_amdgcn_wmma_f32_16x16x32_bf16(
            false, af, false, bf0, (short)0, acc[mi][0], false, false);
        acc[mi][1] = __builtin_amdgcn_wmma_f32_16x16x32_bf16(
            false, af, false, bf1, (short)0, acc[mi][1], false, false);
      }
      __syncthreads();
    }
    // weighted reduce over n inside the fragment (no V materialization)
#pragma unroll
    for (int mi = 0; mi < 4; ++mi) {
      int rb = wm * 64 + mi * 16 + ((lane >> 4) << 3);
#pragma unroll
      for (int ni = 0; ni < 2; ++ni) {
        int col = wn * 32 + ni * 16 + (lane & 15);
        float p = 0.0f;
#pragma unroll
        for (int v = 0; v < 8; ++v)
          p += acc[mi][ni][v] *
               Aout[((size_t)(n0 + rb + v) * 128 + b) * 1024 + c0 + col];
        p += __shfl_xor(p, 16, 32);               // fold lane L with L+16 (same col)
        if (lane < 16) yred[wm][col] += p;        // unique writer per (wm,col)
      }
    }
    __syncthreads();
  }
  if (tid < 128)
    y[(size_t)b * 1024 + c0 + tid] = yred[0][tid] + yred[1][tid] + biasc[c0 + tid];
}

// ------------------------------- prep kernels -------------------------------
__global__ void bn1_cast_kernel(const float* __restrict__ x, const float* __restrict__ g,
                                const float* __restrict__ bt, __bf16* __restrict__ xb, int n) {
  const float inv = rsqrtf(1.0f + 1e-5f);
  for (int i = blockIdx.x * blockDim.x + threadIdx.x; i < n; i += gridDim.x * blockDim.x) {
    int c = i & 255;
    xb[i] = (__bf16)(x[i] * (g[c] * inv) + bt[c]);
  }
}
__global__ void cast_kernel(const float* __restrict__ s, __bf16* __restrict__ d, int n) {
  for (int i = blockIdx.x * blockDim.x + threadIdx.x; i < n; i += gridDim.x * blockDim.x)
    d[i] = (__bf16)s[i];
}
__global__ void ow2_kernel(const float* __restrict__ out_w, const float* __restrict__ g2,
                           __bf16* __restrict__ ow2, int n) {
  const float inv = rsqrtf(1.0f + 1e-5f);
  for (int i = blockIdx.x * blockDim.x + threadIdx.x; i < n; i += gridDim.x * blockDim.x) {
    int d = i & 255;
    ow2[i] = (__bf16)(out_w[i] * g2[d] * inv);
  }
}
__global__ void biasc_kernel(const float* __restrict__ out_w, const float* __restrict__ b2,
                             float* __restrict__ biasc) {
  int c = blockIdx.x * blockDim.x + threadIdx.x;
  if (c < 1024) {
    float s = 0.0f;
    for (int d = 0; d < 256; ++d) s += b2[d] * out_w[(size_t)c * 256 + d];
    biasc[c] = s;
  }
}

// ---------------------------------------------------------------------------
extern "C" void kernel_launch(void* const* d_in, const int* in_sizes, int n_in,
                              void* d_out, int out_size, void* d_ws, size_t ws_size,
                              hipStream_t stream) {
  const float* x    = (const float*)d_in[0];
  const float* WihF = (const float*)d_in[1];
  const float* WhhF = (const float*)d_in[2];
  const float* bihF = (const float*)d_in[3];
  const float* bhhF = (const float*)d_in[4];
  const float* WihB = (const float*)d_in[5];
  const float* WhhB = (const float*)d_in[6];
  const float* bihB = (const float*)d_in[7];
  const float* bhhB = (const float*)d_in[8];
  const float* W_U  = (const float*)d_in[9];
  const float* b_U  = (const float*)d_in[10];
  const float* outw = (const float*)d_in[11];
  const float* bn1g = (const float*)d_in[12];
  const float* bn1b = (const float*)d_in[13];
  const float* bn2g = (const float*)d_in[14];
  const float* bn2b = (const float*)d_in[15];

  float* y = (float*)d_out;                    // (128,1024)
  float* A = (float*)d_out + 131072;           // (256*128,1024): logits -> softmax in place

  char* ws = (char*)d_ws;
  size_t off = 0;
  auto take = [&](size_t bytes) -> char* {
    char* p = ws + off;
    off = (off + bytes + 255) & ~(size_t)255;
    return p;
  };
  __bf16* xb    = (__bf16*)take(8388608ull * 2);    // bn1(x), bf16
  __bf16* wihFb = (__bf16*)take(98304ull * 2);
  __bf16* wihBb = (__bf16*)take(98304ull * 2);
  __bf16* whhFb = (__bf16*)take(49152ull * 2);      // Whh row-major, bf16
  __bf16* whhBb = (__bf16*)take(49152ull * 2);
  __bf16* wUb   = (__bf16*)take(262144ull * 2);
  __bf16* ow2   = (__bf16*)take(262144ull * 2);     // out_w * bn2 scale
  float*  biasc = (float*)take(1024ull * 4);
  __bf16* giF   = (__bf16*)take(12582912ull * 2);   // (128,256,384)
  __bf16* giB   = (__bf16*)take(12582912ull * 2);
  __bf16* Hbf   = (__bf16*)take(8388608ull * 2);    // (256*128,256)

  dim3 blk(256);
  // 1-2) prep
  bn1_cast_kernel<<<4096, blk, 0, stream>>>(x, bn1g, bn1b, xb, 8388608);
  cast_kernel<<<384, blk, 0, stream>>>(WihF, wihFb, 98304);
  cast_kernel<<<384, blk, 0, stream>>>(WihB, wihBb, 98304);
  cast_kernel<<<192, blk, 0, stream>>>(WhhF, whhFb, 49152);
  cast_kernel<<<192, blk, 0, stream>>>(WhhB, whhBb, 49152);
  cast_kernel<<<1024, blk, 0, stream>>>(W_U, wUb, 262144);
  ow2_kernel<<<1024, blk, 0, stream>>>(outw, bn2g, ow2, 262144);
  biasc_kernel<<<4, blk, 0, stream>>>(outw, bn2b, biasc);
  // 3) gi GEMMs: (32768 x 384) = (32768 x 256) x (256 x 384)
  wmma_gemm_bt<__bf16><<<dim3(256, 3), blk, 0, stream>>>(xb, wihFb, bihF, giF, 32768, 384, 256);
  wmma_gemm_bt<__bf16><<<dim3(256, 3), blk, 0, stream>>>(xb, wihBb, bihB, giB, 32768, 384, 256);
  // 4) GRU scan (8 batch chunks x 2 directions), 155648 B dynamic LDS
  size_t lds = 384 * 128 * 2 + 32 * 128 * 2 + 32 * 384 * 4;
  gru_scan_kernel<<<dim3(8, 2), blk, lds, stream>>>(giF, giB, whhFb, whhBb, bhhF, bhhB, Hbf);
  // 5) logits = H x W_U^T + b_U directly into A slot of d_out
  wmma_gemm_bt<float><<<dim3(256, 8), blk, 0, stream>>>(Hbf, wUb, b_U, A, 32768, 1024, 256);
  // 6) softmax over b, in place
  softmax_b_kernel<<<1024, blk, 0, stream>>>(A);
  // 7) fused y
  attn_out_kernel<<<dim3(8, 128), blk, 0, stream>>>(Hbf, ow2, A, biasc, y);
}
